// LSTMCell_89867895701841
// MI455X (gfx1250) — matrix operand order repaired and checked by reading, not compiled, not run on hardware
//
#include <hip/hip_runtime.h>
#include <cstdint>
#include <cstddef>

// ---------------------------------------------------------------------------
// Fused LSTM cell for MI455X (gfx1250, wave32, WMMA).
//   out4 = [x|h] @ [W_ih|W_hh]^T + b   (M=4096, N=8192, K=4096)
//   gates -> c_next, h_next, go  (fused epilogue, no [B,4H] intermediate)
// GEMM on v_wmma_f32_16x16x32_bf16; fp32->bf16 via one v_perm_b32 per pair,
// packed immediately at the load site to minimize live registers.
// 5 per-lane pointers total (1 A + 4 W), stepped once per 4 K-steps; all
// other addressing is compile-time immediate offsets.
// ---------------------------------------------------------------------------

typedef __bf16 bf16_t;
typedef bf16_t v16bf __attribute__((ext_vector_type(16)));
typedef float  v8f   __attribute__((ext_vector_type(8)));

#define B_SZ 4096
#define I_SZ 2048
#define H_SZ 2048
#define K_SZ (I_SZ + H_SZ)      // 4096
#define MT 128                  // rows per block
#define NT 32                   // h-columns per block (x4 gates = 128 N cols)
#define KT 32                   // K per WMMA step
#define LDSS 40                 // LDS row stride in bf16 (KT + 8 pad)
#define KB (KT * 4)             // bytes per K-step along a row (128)
#define AROWB (32 * I_SZ * 4)   // byte stride between a thread's A-load rows (256 KB)

// Pack two fp32 into a bf16x2 dword with ONE VALU op.
__device__ __forceinline__ unsigned pack_bf16x2(float a, float b) {
#if __has_builtin(__builtin_amdgcn_cvt_pk_bf16_f32)
  typedef bf16_t v2bf __attribute__((ext_vector_type(2)));
  union { v2bf v; unsigned u; } r;
  r.v = __builtin_amdgcn_cvt_pk_bf16_f32(a, b);   // lo = a, hi = b
  return r.u;
#else
  union { float f; unsigned u; } ua, ub;
  ua.f = a; ub.f = b;
  return __builtin_amdgcn_perm(ub.u, ua.u, 0x07060302u);  // {b.hi16, a.hi16}
#endif
}

__device__ __forceinline__ uint2 pack_f4(float4 s) {
  uint2 p;
  p.x = pack_bf16x2(s.x, s.y);
  p.y = pack_bf16x2(s.z, s.w);
  return p;
}

union FragU { uint4 q[2]; v16bf v; };

// A fragment (16x32 bf16): lanes 0-15 -> K {0..7},{16..23}; lanes 16-31 -> K {8..15},{24..31}
__device__ __forceinline__ v16bf load_frag_a(const unsigned short* row, int hi) {
  FragU u;
  u.q[0] = *reinterpret_cast<const uint4*>(row + (hi ? 8  : 0));
  u.q[1] = *reinterpret_cast<const uint4*>(row + (hi ? 24 : 16));
  return u.v;
}

// B fragment (32x16 bf16): lanes 0-15 -> K {0..7},{16..23}; lanes 16-31 -> K {8..15},{24..31} (lane = N)
__device__ __forceinline__ v16bf load_frag_b(const unsigned short* row, int hi) {
  FragU u;
  u.q[0] = *reinterpret_cast<const uint4*>(row + (hi ? 8  : 0));
  u.q[1] = *reinterpret_cast<const uint4*>(row + (hi ? 24 : 16));
  return u.v;
}

__device__ __forceinline__ float fast_sigmoid(float x) {
  return 1.0f / (1.0f + __expf(-x));
}
__device__ __forceinline__ float fast_tanh(float x) {
  float e = __expf(-2.0f * x);
  return (1.0f - e) / (1.0f + e);
}

__global__ __launch_bounds__(256)
void lstm_cell_wmma(const float* __restrict__ x, const float* __restrict__ hin,
                    const float* __restrict__ cin,
                    const float* __restrict__ w_ih, const float* __restrict__ w_hh,
                    const float* __restrict__ bias, float* __restrict__ out) {
  // Double-buffered bf16 staging: A = 2*128*40*2B = 20 KB, B same -> 40 KB LDS
  __shared__ unsigned short a_s[2][MT][LDSS];
  __shared__ unsigned short b_s[2][4 * NT][LDSS];

  const int t  = threadIdx.x;
  const int n0 = blockIdx.x * NT;    // h-column tile
  const int m0 = blockIdx.y * MT;    // batch-row tile

  const int lane = t & 31;
  const int wave = t >> 5;           // 0..7
  const int wm   = wave & 3;         // 4 M-groups of 32 rows
  const int wn   = wave >> 2;        // 2 N-groups of 16 cols
  const int hi   = lane >> 4;        // half-wave select
  const int l15  = lane & 15;

  v8f acc[2][4] = {};                // [m-subtile][gate], 16x16 f32 each

  // ---- per-thread base offsets (row = t>>3, float4 index = t&7) ----
  const int trow = t >> 3;           // 0..31
  const int tkq  = t & 7;
  const unsigned a_goff0 =
      (unsigned)((m0 + trow) * I_SZ + tkq * 4) * 4u;           // A base bytes
  // W row for j-th B chunk: j*2048 + n0 + trow  (j folded into 4 pointers)
  unsigned b_goff0[4];
#pragma unroll
  for (int j = 0; j < 4; ++j)
    b_goff0[j] = (unsigned)((j * H_SZ + n0 + trow) * I_SZ + tkq * 4) * 4u;
  // Single LDS staging offset (elements); j-step = 32*LDSS (immediate)
  const unsigned lds_off = (unsigned)(trow * LDSS + tkq * 4);

  // 5 per-lane pointers, re-based once per K-half, stepped 1x per 4 K-steps.
  const char* pA;
  const char* pW[4];
  auto init_ptrs = [&](const float* ap, const float* wp) {
    pA = (const char*)ap + a_goff0;
#pragma unroll
    for (int j = 0; j < 4; ++j) pW[j] = (const char*)wp + b_goff0[j];
  };
  auto advance = [&](int bytes) {
    pA += bytes;
#pragma unroll
    for (int j = 0; j < 4; ++j) pW[j] += bytes;
  };

  uint2 pra[4], prb[4];              // staged, already bf16-packed

  auto load_imm = [&](int u) {       // u compile-time at call site
#pragma unroll
    for (int j = 0; j < 4; ++j)
      pra[j] = pack_f4(*reinterpret_cast<const float4*>(pA + j * AROWB + u * KB));
#pragma unroll
    for (int j = 0; j < 4; ++j)
      prb[j] = pack_f4(*reinterpret_cast<const float4*>(pW[j] + u * KB));
  };

  auto store_tile = [&](int buf) {
    unsigned short* ab = &a_s[buf][0][0];
    unsigned short* bb = &b_s[buf][0][0];
#pragma unroll
    for (int j = 0; j < 4; ++j)
      *reinterpret_cast<uint2*>(ab + lds_off + j * (32 * LDSS)) = pra[j];
#pragma unroll
    for (int j = 0; j < 4; ++j)
      *reinterpret_cast<uint2*>(bb + lds_off + j * (32 * LDSS)) = prb[j];
  };

  auto compute = [&](int buf) {
    const unsigned short* ab = &a_s[buf][0][0];
    const unsigned short* bb = &b_s[buf][0][0];
    const v16bf af0 = load_frag_a(ab + (wm * 32 + l15) * LDSS, hi);
    const v16bf af1 = load_frag_a(ab + (wm * 32 + 16 + l15) * LDSS, hi);
#pragma unroll
    for (int g = 0; g < 4; ++g) {
      const v16bf bf = load_frag_b(bb + (g * NT + wn * 16 + l15) * LDSS, hi);
      acc[0][g] = __builtin_amdgcn_wmma_f32_16x16x32_bf16(
          false, af0, false, bf, (short)0, acc[0][g], false, false);
      acc[1][g] = __builtin_amdgcn_wmma_f32_16x16x32_bf16(
          false, af1, false, bf, (short)0, acc[1][g], false, false);
    }
  };

  // ---- software pipeline: windows of 4 K-steps, double-buffered LDS ----
  init_ptrs(x, w_ih);
  load_imm(0);                       // tile 0
  store_tile(0);
  __syncthreads();
  advance(KB);                       // pointers -> tile 1

  // Half 0: computes tiles 0..63 (x @ w_ih); loads run one tile ahead.
  for (int w = 0; w < 16; ++w) {
    const bool last = (w == 15);
    load_imm(0); compute(0); store_tile(1); __syncthreads();
    load_imm(1); compute(1); store_tile(0); __syncthreads();
    load_imm(2); compute(0); store_tile(1); __syncthreads();
    if (last) { init_ptrs(hin, w_hh); load_imm(0); }  // tile 64 = h-half k=0
    else      { load_imm(3); }
    compute(1); store_tile(0); __syncthreads();
    advance(last ? KB : 4 * KB);
  }

  // Half 1: computes tiles 64..127 (h @ w_hh).
  for (int w = 0; w < 16; ++w) {
    const bool last = (w == 15);
    load_imm(0); compute(0); store_tile(1); __syncthreads();
    load_imm(1); compute(1); store_tile(0); __syncthreads();
    load_imm(2); compute(0); store_tile(1); __syncthreads();
    if (!last) load_imm(3);
    compute(1);
    if (!last) store_tile(0);
    __syncthreads();
    advance(4 * KB);
  }

  // ---- fused LSTM epilogue ----
  // C/D layout: lane 0-15 -> N=lane, M=v; lane 16-31 -> N=lane-16, M=v+8
  const int n = n0 + wn * 16 + l15;
  const float b0 = bias[0 * H_SZ + n];
  const float b1 = bias[1 * H_SZ + n];
  const float b2 = bias[2 * H_SZ + n];
  const float b3 = bias[3 * H_SZ + n];
  const size_t BH = (size_t)B_SZ * H_SZ;

#pragma unroll
  for (int i = 0; i < 2; ++i) {
#pragma unroll
    for (int v = 0; v < 8; ++v) {
      const int m = m0 + wm * 32 + i * 16 + v + hi * 8;
      const size_t idx = (size_t)m * H_SZ + n;
      const float gf = fast_sigmoid(acc[i][0][v] + b0);
      const float gi = fast_sigmoid(acc[i][1][v] + b1);
      const float gs = fast_tanh   (acc[i][2][v] + b2);
      const float go = fast_sigmoid(acc[i][3][v] + b3);
      const float cn = gf * cin[idx] + gi * gs;
      const float hn = go * fast_tanh(cn);
      out[idx]          = hn;   // h_next
      out[BH + idx]     = cn;   // c_next
      out[2 * BH + idx] = go;   // go
    }
  }
}

extern "C" void kernel_launch(void* const* d_in, const int* in_sizes, int n_in,
                              void* d_out, int out_size, void* d_ws, size_t ws_size,
                              hipStream_t stream) {
  (void)in_sizes; (void)n_in; (void)out_size; (void)d_ws; (void)ws_size;
  const float* x    = (const float*)d_in[0];
  const float* h    = (const float*)d_in[1];
  const float* c    = (const float*)d_in[2];
  // d_in[3] = "o" is unused by the reference computation
  const float* w_ih = (const float*)d_in[4];
  const float* w_hh = (const float*)d_in[5];
  const float* bias = (const float*)d_in[6];
  float* out = (float*)d_out;

  dim3 grid(H_SZ / NT, B_SZ / MT);   // (64, 32) = 2048 blocks
  lstm_cell_wmma<<<grid, dim3(256), 0, stream>>>(x, h, c, w_ih, w_hh, bias, out);
}
